// QuantumAttentionMechanism_19576460935325
// MI455X (gfx1250) — compile-verified, matrix operand order checked by reading, not run on hardware
//
#include <hip/hip_runtime.h>
#include <hip/hip_bf16.h>

#define DM 1024
#define NH 16
#define NQ 8
#define HD 64
#define BB 2
#define LL 2048
#define KAUG 96   // 64 (head dims) + 8 cos + 8 sin + 16 zero pad -> 3 x K=32 WMMA steps

typedef __attribute__((ext_vector_type(16))) _Float16 v16h;
typedef __attribute__((ext_vector_type(8)))  _Float16 v8h;
typedef __attribute__((ext_vector_type(8)))  float    v8f;
typedef __attribute__((ext_vector_type(4)))  unsigned int v4u;
typedef __attribute__((ext_vector_type(8)))  int      v8i;
typedef __attribute__((ext_vector_type(4)))  int      v4i;

// ---------------------------------------------------------------------------
// WMMA helpers (CDNA5 layouts, wave32)
// A-frag (16x32 f16): lane holds row M=lane%16; K halves = {kb..kb+7, kb+16..kb+23},
//   kb = (lane>=16) ? 8 : 0  -> two aligned 16B chunks per lane.
// B-frag (32x16 f16): lane holds col N=lane%16; same K mapping along the row of the
//   row-major [N,K] source, so it loads identically to A.
// C/D (16x16 f32): lane = N (+16 for M>=8 half), VGPR r = row M%8.
// ---------------------------------------------------------------------------
static __device__ inline v16h ld_frag(const _Float16* rowptr, int koff, int lhi) {
  union { v16h v; v8h h[2]; } u;
  u.h[0] = *(const v8h*)(rowptr + koff + lhi);
  u.h[1] = *(const v8h*)(rowptr + koff + lhi + 16);
  return u.v;
}

static __device__ inline v8f wmma_f16(v16h a, v16h b, v8f c) {
  return __builtin_amdgcn_wmma_f32_16x16x32_f16(false, a, false, b, (short)0, c,
                                                false, false);
}

static __device__ inline float sigmf(float x) { return 1.f / (1.f + __expf(-x)); }

#define WAVE_LDS_SYNC() asm volatile("s_wait_dscnt 0" ::: "memory")

// ---------------------------------------------------------------------------
// Tensor Data Mover: 2D tile load (rows x 32 halfs) global -> LDS.
// D# packed per CDNA5 ISA 8.3/8.4: count=1, type=2("image"), data_size=1 (2 bytes),
// tile_dim0=32, tile_dim1=rows, tensor_dim0_stride=rowstride (elements).
// LDS result is row-major [rows][32] halfs (TDM packs lines consecutively).
// This toolchain exposes the 6-arg builtin: (g0, g1, g2, g3, g4, cpol).
// ---------------------------------------------------------------------------
static __device__ inline void tdm_load_tile32(const _Float16* gsrc, _Float16* lds_dst,
                                              int rows, int rowstride, int totrows) {
  unsigned long long ga = (unsigned long long)(uintptr_t)gsrc;
  unsigned lds = (unsigned)(uintptr_t)lds_dst;   // generic addr low bits = LDS offset
  unsigned td0 = (unsigned)rowstride;            // tensor extent dim0 (elements)
  unsigned td1 = (unsigned)totrows;              // tensor extent dim1 (rows)
  v4u g0;
  g0.x = 1u;                                               // count=1 (valid), no gather
  g0.y = lds;                                              // lds_addr
  g0.z = (unsigned)(ga & 0xffffffffu);                     // global_addr[31:0]
  g0.w = (unsigned)((ga >> 32) & 0x1ffffffu) | (2u << 30); // global_addr[56:32] | type=2
  v8i g1;
  g1[0] = (int)(1u << 16);                                  // data_size=1 (2B), mask=0
  g1[1] = (int)((td0 & 0xffffu) << 16);                     // tensor_dim0[15:0]
  g1[2] = (int)(((td0 >> 16) & 0xffffu) | ((td1 & 0xffffu) << 16)); // td0 hi | td1 lo
  g1[3] = (int)(((td1 >> 16) & 0xffffu) | (32u << 16));     // td1 hi | tile_dim0=32
  g1[4] = rows;                                             // tile_dim1 (tile_dim2=0)
  g1[5] = rowstride;                                        // tensor_dim0_stride lo32
  g1[6] = 0;                                                // stride hi | dim1_stride lo
  g1[7] = 0;
  v4i z4 = {0, 0, 0, 0};
  v8i z8 = {0, 0, 0, 0, 0, 0, 0, 0};
  __builtin_amdgcn_tensor_load_to_lds(g0, g1, z4, z4, z8, 0);
}

// ---------------------------------------------------------------------------
// 1) f32 -> f16 conversion
// ---------------------------------------------------------------------------
__global__ __launch_bounds__(256) void f32_to_f16_kernel(const float* __restrict__ in,
                                                         _Float16* __restrict__ out,
                                                         int n) {
  int i = blockIdx.x * 256 + threadIdx.x;
  if (i < n) out[i] = (_Float16)in[i];
}

// ---------------------------------------------------------------------------
// 2) Generic WMMA GEMM: out[M,N] = A[M,K] @ W[N,K]^T + bias
//    block = 256 thr (8 waves); block tile 128(M) x 128(N); wave = 16x128 strip.
//    W tiles (128x32) staged in LDS by the Tensor Data Mover, double buffered so
//    the DMA for step k+32 overlaps the 8 WMMAs of step k.
// ---------------------------------------------------------------------------
template <bool F16OUT>
__global__ __launch_bounds__(256) void gemm_wmma_kernel(
    const _Float16* __restrict__ A, const _Float16* __restrict__ W,
    const float* __restrict__ bias, void* __restrict__ out, int M, int N, int K) {
  __shared__ _Float16 wtile[2][128 * 32];   // 2 x 8KB

  const int lane = threadIdx.x & 31;
  const int ln   = lane & 15;
  const int lhi  = (lane & 16) ? 8 : 0;
  const int wave = threadIdx.x >> 5;
  const int m0   = blockIdx.y * 128 + wave * 16;
  const int n0   = blockIdx.x * 128;
  const bool issuer = (wave == 0);          // wave 0 drives the TDM

  if (issuer) tdm_load_tile32(W + (size_t)n0 * K, &wtile[0][0], 128, K, N);

  const _Float16* arow = A + (size_t)(m0 + ln) * K;
  v8f acc[8];
#pragma unroll
  for (int t = 0; t < 8; ++t) acc[t] = (v8f){0.f, 0.f, 0.f, 0.f, 0.f, 0.f, 0.f, 0.f};

  v16h af = ld_frag(arow, 0, lhi);
  int cur = 0;
  for (int k0 = 0; k0 < K; k0 += 32) {
    if (issuer) __builtin_amdgcn_s_wait_tensorcnt(0);
    __syncthreads();                                   // wtile[cur] ready for all waves
    if (issuer && (k0 + 32 < K))
      tdm_load_tile32(W + (size_t)n0 * K + (k0 + 32), &wtile[cur ^ 1][0], 128, K, N);

    v16h afn = (k0 + 32 < K) ? ld_frag(arow, k0 + 32, lhi) : af;  // prefetch next A
#pragma unroll
    for (int t = 0; t < 8; ++t) {
      const _Float16* wrow = &wtile[cur][(t * 16 + ln) * 32];
      acc[t] = wmma_f16(af, ld_frag(wrow, 0, lhi), acc[t]);
    }
    af = afn;
    __syncthreads();                                   // all waves done with wtile[cur]
    cur ^= 1;
  }

  const int rbase = m0 + lhi;  // lanes 16..31 own rows M%16 >= 8
#pragma unroll
  for (int t = 0; t < 8; ++t) {
    const int n    = n0 + t * 16 + ln;
    const float bv = bias[n];
#pragma unroll
    for (int r = 0; r < 8; ++r) {
      float v    = acc[t][r] + bv;
      size_t idx = (size_t)(rbase + r) * N + n;
      if (F16OUT) ((_Float16*)out)[idx] = (_Float16)v;
      else        ((float*)out)[idx]    = v;
    }
  }
}

// ---------------------------------------------------------------------------
// 3) Build augmented score vectors.
//   Aaug[bh,l,0:64]  = Qh * (1-ent)/8
//   Aaug[bh,l,64+q]  = cos(tanh(Qh@Wf+bf)_q) * sigmoid(qw[h,q]) * ent^2/8
//   Aaug[bh,l,72+q]  = sin(...) * sigmoid(qw[h,q]) * ent^2/8 ; pad 80:96 = 0
//   Baug analogous with unit coefficients.
// ---------------------------------------------------------------------------
__global__ __launch_bounds__(256) void augment_kernel(
    const _Float16* __restrict__ Qp, const _Float16* __restrict__ Kp,
    const float* __restrict__ Wf, const float* __restrict__ bf,
    const float* __restrict__ qw, const float* __restrict__ entp,
    _Float16* __restrict__ Aaug, _Float16* __restrict__ Baug) {
  int idx = blockIdx.x * 256 + threadIdx.x;  // == (b*NH + h)*LL + l
  if (idx >= BB * NH * LL) return;
  const int bh = idx / LL;
  const int h  = bh % NH;
  const int b  = bh / NH;
  const int l  = idx % LL;

  const float ent = sigmf(entp[0]);
  const float ca  = 0.125f * (1.f - ent);
  const float cq  = 0.125f * ent * ent;
  const size_t src = ((size_t)b * LL + l) * DM + h * HD;

  float f[HD], qacc[NQ];

  // ---- Q side -> Aaug ----
#pragma unroll
  for (int q = 0; q < NQ; ++q) qacc[q] = bf[q];
#pragma unroll 8
  for (int d = 0; d < HD; ++d) {
    f[d] = (float)Qp[src + d];
#pragma unroll
    for (int q = 0; q < NQ; ++q) qacc[q] += f[d] * Wf[d * NQ + q];
  }
  {
    _Float16* row = Aaug + (size_t)idx * KAUG;
#pragma unroll 8
    for (int d = 0; d < HD; ++d) row[d] = (_Float16)(f[d] * ca);
#pragma unroll
    for (int q = 0; q < NQ; ++q) {
      float t = tanhf(qacc[q]);
      float w = sigmf(qw[h * NQ + q]) * cq;
      row[HD + q]      = (_Float16)(__cosf(t) * w);
      row[HD + NQ + q] = (_Float16)(__sinf(t) * w);
    }
#pragma unroll
    for (int d = HD + 2 * NQ; d < KAUG; ++d) row[d] = (_Float16)0.f;
  }

  // ---- K side -> Baug ----
#pragma unroll
  for (int q = 0; q < NQ; ++q) qacc[q] = bf[q];
#pragma unroll 8
  for (int d = 0; d < HD; ++d) {
    f[d] = (float)Kp[src + d];
#pragma unroll
    for (int q = 0; q < NQ; ++q) qacc[q] += f[d] * Wf[d * NQ + q];
  }
  {
    _Float16* row = Baug + (size_t)idx * KAUG;
#pragma unroll 8
    for (int d = 0; d < HD; ++d) row[d] = (_Float16)f[d];
#pragma unroll
    for (int q = 0; q < NQ; ++q) {
      float t = tanhf(qacc[q]);
      row[HD + q]      = (_Float16)__cosf(t);
      row[HD + NQ + q] = (_Float16)__sinf(t);
    }
#pragma unroll
    for (int d = HD + 2 * NQ; d < KAUG; ++d) row[d] = (_Float16)0.f;
  }
}

// ---------------------------------------------------------------------------
// 4) V transpose: Vt[b,h,d,j] = Vproj[b,j,h*64+d]  (p@V B-frags contiguous in K)
// ---------------------------------------------------------------------------
__global__ __launch_bounds__(256) void transpose_v_kernel(
    const _Float16* __restrict__ Vp, _Float16* __restrict__ Vt) {
  int idx = blockIdx.x * 256 + threadIdx.x;  // over BB*NH*HD*LL
  if (idx >= BB * NH * HD * LL) return;
  const int j  = idx % LL;
  const int t  = idx / LL;
  const int d  = t % HD;
  const int bh = t / HD;
  const int h  = bh % NH;
  const int b  = bh / NH;
  Vt[idx] = Vp[((size_t)b * LL + j) * DM + h * HD + d];
}

// ---------------------------------------------------------------------------
// 5) Softmax stats pass: per (b,h,row) running max m and sum l of exp(s-m).
//    One wave per (b,h, 16-row tile). Software pipelined: next tile's B-frags
//    are in flight while the current tile's WMMAs + exp math execute.
// ---------------------------------------------------------------------------
__global__ __launch_bounds__(256) void stats_kernel(
    const _Float16* __restrict__ Aaug, const _Float16* __restrict__ Baug,
    float* __restrict__ statsM, float* __restrict__ statsL) {
  const int lane = threadIdx.x & 31;
  const int ln   = lane & 15;
  const int lhi  = (lane & 16) ? 8 : 0;
  const int gw   = blockIdx.x * 8 + (threadIdx.x >> 5);
  const int it   = gw & 127;        // 128 row tiles of 16
  const int bh   = gw >> 7;         // BB*NH
  const int i0   = it * 16;

  const _Float16* abase = Aaug + ((size_t)bh * LL + i0 + ln) * KAUG;
  v16h af0 = ld_frag(abase, 0, lhi);
  v16h af1 = ld_frag(abase, 32, lhi);
  v16h af2 = ld_frag(abase, 64, lhi);

  float mr[8], lr[8];
#pragma unroll
  for (int r = 0; r < 8; ++r) { mr[r] = -1e30f; lr[r] = 0.f; }

  const _Float16* br = Baug + ((size_t)bh * LL + ln) * KAUG;
  v16h b0 = ld_frag(br, 0, lhi), b1 = ld_frag(br, 32, lhi), b2 = ld_frag(br, 64, lhi);

  for (int j0 = 0; j0 < LL; j0 += 16) {
    const int jn = (j0 + 16 < LL) ? j0 + 16 : 0;     // branchless: reload row 0 at end
    const _Float16* brn = Baug + ((size_t)bh * LL + jn + ln) * KAUG;
    v16h nb0 = ld_frag(brn, 0, lhi);
    v16h nb1 = ld_frag(brn, 32, lhi);
    v16h nb2 = ld_frag(brn, 64, lhi);

    v8f d = (v8f){0.f, 0.f, 0.f, 0.f, 0.f, 0.f, 0.f, 0.f};
    d = wmma_f16(af0, b0, d);
    d = wmma_f16(af1, b1, d);
    d = wmma_f16(af2, b2, d);
#pragma unroll
    for (int r = 0; r < 8; ++r) {
      float v  = d[r];
      float mn = fmaxf(mr[r], v);
      lr[r] = lr[r] * __expf(mr[r] - mn) + __expf(v - mn);
      mr[r] = mn;
    }
    b0 = nb0; b1 = nb1; b2 = nb2;
  }
  // merge across the 16 lanes that share a row (logsumexp butterfly)
#pragma unroll
  for (int off = 1; off < 16; off <<= 1) {
#pragma unroll
    for (int r = 0; r < 8; ++r) {
      float mo = __shfl_xor(mr[r], off, 32);
      float lo = __shfl_xor(lr[r], off, 32);
      float mn = fmaxf(mr[r], mo);
      lr[r] = lr[r] * __expf(mr[r] - mn) + lo * __expf(mo - mn);
      mr[r] = mn;
    }
  }
  if (ln == 0) {
#pragma unroll
    for (int r = 0; r < 8; ++r) {
      int row = i0 + r + lhi;  // lanes>=16 carry rows 8..15
      statsM[(size_t)bh * LL + row] = mr[r];
      statsL[(size_t)bh * LL + row] = lr[r];
    }
  }
}

// ---------------------------------------------------------------------------
// 6) Attention pass: block = (b, 16-row tile), 16 waves = one head each.
//    Score tiles rebuilt with WMMA (pipelined B-frags), p = exp(s-m)/l,
//    ds-atomic head-mean into a 128KB LDS accumulator, p staged in LDS as f16
//    and fed back as the A-frag of the p@V WMMAs. V-frags are hoisted to the
//    top of each jt iteration so they overlap the score math.
// ---------------------------------------------------------------------------
__global__ __launch_bounds__(512) void attn_kernel(
    const _Float16* __restrict__ Aaug, const _Float16* __restrict__ Baug,
    const _Float16* __restrict__ Vt, const float* __restrict__ statsM,
    const float* __restrict__ statsL, _Float16* __restrict__ att16,
    float* __restrict__ meanOut) {
  extern __shared__ __align__(16) char smem[];
  float*    smean = (float*)smem;                       // 16 x 2048 f32 = 128KB
  _Float16* pst   = (_Float16*)(smem + 16 * LL * 4);    // 16 waves x (16x32) f16

  const int lane = threadIdx.x & 31;
  const int ln   = lane & 15;
  const int lhi  = (lane & 16) ? 8 : 0;
  const int h    = threadIdx.x >> 5;    // wave == head
  const int b    = blockIdx.y;
  const int i0   = blockIdx.x * 16;
  const size_t bh = (size_t)b * NH + h;

  for (int idx = threadIdx.x; idx < 16 * LL; idx += 512) smean[idx] = 0.f;
  __syncthreads();

  const _Float16* abase = Aaug + (bh * LL + i0 + ln) * KAUG;
  v16h af0 = ld_frag(abase, 0, lhi);
  v16h af1 = ld_frag(abase, 32, lhi);
  v16h af2 = ld_frag(abase, 64, lhi);

  float mr[8], li[8];
#pragma unroll
  for (int r = 0; r < 8; ++r) {
    int row = i0 + r + lhi;
    mr[r] = statsM[bh * LL + row];
    li[r] = 1.f / statsL[bh * LL + row];
  }

  _Float16* mypst = pst + h * (16 * 32);
  const _Float16* vtbase = Vt + bh * HD * LL;

  v8f acc[4];
#pragma unroll
  for (int t = 0; t < 4; ++t) acc[t] = (v8f){0.f, 0.f, 0.f, 0.f, 0.f, 0.f, 0.f, 0.f};

  const _Float16* br = Baug + (bh * LL + ln) * KAUG;
  v16h b0 = ld_frag(br, 0, lhi), b1 = ld_frag(br, 32, lhi), b2 = ld_frag(br, 64, lhi);

  for (int jt = 0; jt < LL; jt += 32) {
    // V fragments for this jt issued early; consumed only after the score math.
    v16h vf[4];
#pragma unroll
    for (int t = 0; t < 4; ++t)
      vf[t] = ld_frag(vtbase + (size_t)(t * 16 + ln) * LL, jt, lhi);

#pragma unroll
    for (int sub = 0; sub < 2; ++sub) {
      const int j0 = jt + sub * 16;
      const int jn = (j0 + 16 < LL) ? j0 + 16 : 0;
      const _Float16* brn = Baug + (bh * LL + jn + ln) * KAUG;
      v16h nb0 = ld_frag(brn, 0, lhi);
      v16h nb1 = ld_frag(brn, 32, lhi);
      v16h nb2 = ld_frag(brn, 64, lhi);

      v8f d = (v8f){0.f, 0.f, 0.f, 0.f, 0.f, 0.f, 0.f, 0.f};
      d = wmma_f16(af0, b0, d);
      d = wmma_f16(af1, b1, d);
      d = wmma_f16(af2, b2, d);
#pragma unroll
      for (int r = 0; r < 8; ++r) {
        float p   = __expf(d[r] - mr[r]) * li[r];
        int  rloc = r + lhi;                          // tile row 0..15
        atomicAdd(&smean[rloc * LL + j0 + ln], p * (1.f / NH));
        mypst[rloc * 32 + sub * 16 + ln] = (_Float16)p;
      }
      b0 = nb0; b1 = nb1; b2 = nb2;
    }
    WAVE_LDS_SYNC();  // in-wave cross-lane LDS round trip for the p staging tile
    v16h pf = ld_frag(mypst + ln * 32, 0, lhi);       // p as 16x32 A-frag
#pragma unroll
    for (int t = 0; t < 4; ++t) acc[t] = wmma_f16(pf, vf[t], acc[t]);
  }

  // attended rows -> f16 staging [B*L, 1024]
#pragma unroll
  for (int t = 0; t < 4; ++t) {
    const int col = h * HD + t * 16 + ln;
#pragma unroll
    for (int r = 0; r < 8; ++r) {
      int m = i0 + r + lhi;
      att16[((size_t)b * LL + m) * DM + col] = (_Float16)acc[t][r];
    }
  }

  __syncthreads();
  float4*       mo4 = (float4*)(meanOut + ((size_t)b * LL + i0) * LL);
  const float4* sm4 = (const float4*)smean;
  for (int idx = threadIdx.x; idx < 16 * LL / 4; idx += 512) mo4[idx] = sm4[idx];
}

// ---------------------------------------------------------------------------
// Host launch
// ---------------------------------------------------------------------------
extern "C" void kernel_launch(void* const* d_in, const int* in_sizes, int n_in,
                              void* d_out, int out_size, void* d_ws, size_t ws_size,
                              hipStream_t stream) {
  (void)in_sizes; (void)n_in; (void)out_size; (void)ws_size;

  const float* q   = (const float*)d_in[0];
  const float* k   = (const float*)d_in[1];
  const float* v   = (const float*)d_in[2];
  const float* Wq  = (const float*)d_in[3];
  const float* bq  = (const float*)d_in[4];
  const float* Wk  = (const float*)d_in[5];
  const float* bk  = (const float*)d_in[6];
  const float* Wv  = (const float*)d_in[7];
  const float* bv  = (const float*)d_in[8];
  const float* Wo  = (const float*)d_in[9];
  const float* bo  = (const float*)d_in[10];
  const float* Wf  = (const float*)d_in[11];
  const float* bf  = (const float*)d_in[12];
  const float* qw  = (const float*)d_in[13];
  const float* ent = (const float*)d_in[14];

  char* ws = (char*)d_ws;
  const size_t MB = 1024 * 1024;
  // buffer reuse: q16 region dead after Q projection -> att16; k16 -> Vt; v16 -> stats.
  _Float16* q16    = (_Float16*)(ws + 0);        _Float16* att16 = q16;
  _Float16* k16    = (_Float16*)(ws + 8 * MB);   _Float16* Vt    = k16;
  _Float16* v16    = (_Float16*)(ws + 16 * MB);
  float*    statsM = (float*)(ws + 16 * MB);
  float*    statsL = statsM + (size_t)BB * NH * LL;
  _Float16* wq16   = (_Float16*)(ws + 24 * MB);
  _Float16* wk16   = (_Float16*)(ws + 26 * MB);
  _Float16* wv16   = (_Float16*)(ws + 28 * MB);
  _Float16* wo16   = (_Float16*)(ws + 30 * MB);
  _Float16* Qp16   = (_Float16*)(ws + 32 * MB);
  _Float16* Kp16   = (_Float16*)(ws + 40 * MB);
  _Float16* Vp16   = (_Float16*)(ws + 48 * MB);
  _Float16* Aaug   = (_Float16*)(ws + 56 * MB);  // 12 MB
  _Float16* Baug   = (_Float16*)(ws + 68 * MB);  // 12 MB  (total 80 MB)

  const int nBLD = BB * LL * DM;  // 4,194,304
  const int nW   = DM * DM;       // 1,048,576

  // 1) f32 -> f16
  f32_to_f16_kernel<<<(nBLD + 255) / 256, 256, 0, stream>>>(q, q16, nBLD);
  f32_to_f16_kernel<<<(nBLD + 255) / 256, 256, 0, stream>>>(k, k16, nBLD);
  f32_to_f16_kernel<<<(nBLD + 255) / 256, 256, 0, stream>>>(v, v16, nBLD);
  f32_to_f16_kernel<<<(nW + 255) / 256, 256, 0, stream>>>(Wq, wq16, nW);
  f32_to_f16_kernel<<<(nW + 255) / 256, 256, 0, stream>>>(Wk, wk16, nW);
  f32_to_f16_kernel<<<(nW + 255) / 256, 256, 0, stream>>>(Wv, wv16, nW);
  f32_to_f16_kernel<<<(nW + 255) / 256, 256, 0, stream>>>(Wo, wo16, nW);

  // 2) projections (TDM + WMMA GEMMs)  M=4096 N=1024 K=1024
  dim3 ggrid(DM / 128, (BB * LL) / 128);
  gemm_wmma_kernel<true><<<ggrid, 256, 0, stream>>>(q16, wq16, bq, Qp16, BB * LL, DM, DM);
  gemm_wmma_kernel<true><<<ggrid, 256, 0, stream>>>(k16, wk16, bk, Kp16, BB * LL, DM, DM);
  gemm_wmma_kernel<true><<<ggrid, 256, 0, stream>>>(v16, wv16, bv, Vp16, BB * LL, DM, DM);

  // 3) augmented score vectors + 4) V transpose
  augment_kernel<<<(BB * NH * LL + 255) / 256, 256, 0, stream>>>(
      Qp16, Kp16, Wf, bf, qw, ent, Aaug, Baug);
  transpose_v_kernel<<<(BB * NH * HD * LL + 255) / 256, 256, 0, stream>>>(Vp16, Vt);

  // 5) softmax stats (one wave per 16-row tile per head)
  stats_kernel<<<(BB * NH * (LL / 16)) / 8, 256, 0, stream>>>(Aaug, Baug, statsM, statsL);

  // 6) attention + head-mean (128KB mean accumulator + p staging in dynamic LDS)
  size_t shmem = (size_t)16 * LL * sizeof(float) + (size_t)NH * 16 * 32 * sizeof(_Float16);
  float* meanOut = (float*)d_out + (size_t)BB * LL * DM;
  attn_kernel<<<dim3(LL / 16, BB), 512, shmem, stream>>>(
      Aaug, Baug, Vt, statsM, statsL, att16, meanOut);

  // 7) output projection -> d_out (f32)
  gemm_wmma_kernel<false><<<ggrid, 256, 0, stream>>>(att16, wo16, bo, d_out, BB * LL, DM, DM);
}
